// LSTMcell_8804682957036
// MI455X (gfx1250) — compile-verified, matrix operand order checked by reading, not run on hardware
//
#include <hip/hip_runtime.h>
#include <hip/hip_bf16.h>

// ---------------------------------------------------------------------------
// LSTM cell, MI455X (gfx1250).  bf16 WMMA GEMM (fp32 accum) + fused epilogue.
//   gates[b, g*D+n] = sum_k A[b,k] * Wp[g*D+n, k] + bsum[g*D+n]
//   A  = [x | h]            (bf16, [B, 2D])
//   Wp = [W_g_x | W_g_h]    (bf16, [4D, 2D]),  g in {f,i,o,c}
// ---------------------------------------------------------------------------

#define Dn   2048
#define Bn   8192
#define K2   4096   // 2*Dn

typedef __attribute__((ext_vector_type(16))) __bf16 v16bf;
typedef __attribute__((ext_vector_type(8)))  __bf16 v8bf;
typedef __attribute__((ext_vector_type(8)))  float  v8f;

#define WMMA_BF16(a, b, c) \
    __builtin_amdgcn_wmma_f32_16x16x32_bf16(false, (a), false, (b), (short)0, (c), false, false)

__device__ __forceinline__ unsigned short f2bf_rne(float f) {
    unsigned u = __float_as_uint(f);
    u += 0x7FFFu + ((u >> 16) & 1u);            // round-to-nearest-even
    return (unsigned short)(u >> 16);
}

__device__ __forceinline__ float sigm(float x)  { return 1.0f / (1.0f + __expf(-x)); }
__device__ __forceinline__ float tanhp(float x) { return 2.0f / (1.0f + __expf(-2.0f * x)) - 1.0f; }

// ---------------------------------------------------------------------------
// Packing kernels (fp32 -> bf16, layout rearrangement).  4 elems / thread.
// ---------------------------------------------------------------------------
__global__ void pack_a_kernel(const float* __restrict__ x, const float* __restrict__ h,
                              unsigned short* __restrict__ A) {
    size_t idx = ((size_t)blockIdx.x * blockDim.x + threadIdx.x) * 4;
    size_t row = idx >> 12;                     // / 4096
    int    k   = (int)(idx & 4095);
    const float* src = (k < Dn) ? (x + row * Dn + k) : (h + row * Dn + (k - Dn));
    float4 v = *(const float4*)src;
    ushort4 o;
    o.x = f2bf_rne(v.x); o.y = f2bf_rne(v.y); o.z = f2bf_rne(v.z); o.w = f2bf_rne(v.w);
    *(ushort4*)(A + idx) = o;
}

__global__ void pack_w_kernel(const float* __restrict__ Wfx, const float* __restrict__ Wfh,
                              const float* __restrict__ Wix, const float* __restrict__ Wih,
                              const float* __restrict__ Wox, const float* __restrict__ Woh,
                              const float* __restrict__ Wcx, const float* __restrict__ Wch,
                              unsigned short* __restrict__ Wp) {
    size_t idx = ((size_t)blockIdx.x * blockDim.x + threadIdx.x) * 4;
    size_t row = idx >> 12;                     // gate-major row in [0, 4*Dn)
    int    k   = (int)(idx & 4095);
    int    g   = (int)(row >> 11);
    size_t n   = row & 2047;
    const float* W;
    if (k < Dn) W = (g == 0) ? Wfx : (g == 1) ? Wix : (g == 2) ? Wox : Wcx;
    else        W = (g == 0) ? Wfh : (g == 1) ? Wih : (g == 2) ? Woh : Wch;
    const float* src = W + n * Dn + (k < Dn ? k : k - Dn);
    float4 v = *(const float4*)src;
    ushort4 o;
    o.x = f2bf_rne(v.x); o.y = f2bf_rne(v.y); o.z = f2bf_rne(v.z); o.w = f2bf_rne(v.w);
    *(ushort4*)(Wp + idx) = o;
}

__global__ void pack_bias_kernel(const float* __restrict__ bfx, const float* __restrict__ bfh,
                                 const float* __restrict__ bix, const float* __restrict__ bih,
                                 const float* __restrict__ box_, const float* __restrict__ boh,
                                 const float* __restrict__ bcx, const float* __restrict__ bch,
                                 float* __restrict__ bsum) {
    int i = blockIdx.x * blockDim.x + threadIdx.x;   // [0, 4*Dn)
    int g = i >> 11;
    int n = i & 2047;
    const float* bx = (g == 0) ? bfx : (g == 1) ? bix : (g == 2) ? box_ : bcx;
    const float* bh = (g == 0) ? bfh : (g == 1) ? bih : (g == 2) ? boh : bch;
    bsum[i] = bx[n] + bh[n];
}

// ---------------------------------------------------------------------------
// Fragment loader: ISA 16-bit A/B VGPR layout for 16x16x32 WMMA.
// lane<16 : rows r=lane,    K = {k0..k0+7, k0+16..k0+23}
// lane>=16: rows r=lane-16, K = {k0+8..k0+15, k0+24..k0+31}
// (same pattern for B since lane<->N, regs<->K; W is stored K-major per row)
// ---------------------------------------------------------------------------
__device__ __forceinline__ v16bf load_frag(const unsigned short* __restrict__ base,
                                           unsigned off /* element offset */) {
    const unsigned short* p = base + off;
    v8bf lo = *(const v8bf*)(p);
    v8bf hi = *(const v8bf*)(p + 16);
    v16bf f;
#pragma unroll
    for (int i = 0; i < 8; ++i) { f[i] = lo[i]; f[i + 8] = hi[i]; }
    return f;
}

// ---------------------------------------------------------------------------
// Fused GEMM + LSTM epilogue.
// Grid: (Dn/16, Bn/256).  Block: 256 threads = 8 waves.
// Wave: 32 rows x 16 cols, all 4 gates -> 8 f32 accumulators (64 VGPRs).
// sched_barrier(VMEM_READ) pins "all 12 loads, then 8 WMMAs" per iteration:
// wait-insertion emits decreasing partial s_wait_loadcnt, and next-iteration
// loads may still hoist across the barrier for cross-iteration pipelining.
// ---------------------------------------------------------------------------
__global__ __launch_bounds__(256)
void lstm_gemm_fused(const unsigned short* __restrict__ A,
                     const unsigned short* __restrict__ W,
                     const float* __restrict__ bsum,
                     const float* __restrict__ c_in,
                     float* __restrict__ out_o,
                     float* __restrict__ out_h,
                     float* __restrict__ out_c) {
    const int lane   = threadIdx.x & 31;
    const int wave   = threadIdx.x >> 5;
    const int n_base = blockIdx.x << 4;                       // 16 output cols
    const int m0     = (blockIdx.y << 8) + (wave << 5);       // 32 rows / wave

    const int rsel = lane & 15;
    const int ksel = (lane >> 4) << 3;                        // 0 or 8

    // 32-bit per-lane element offsets against uniform bases (A, W < 2^31 elems)
    const unsigned offA0 = (unsigned)(m0 + rsel) * K2 + ksel;
    const unsigned offA1 = offA0 + 16u * K2;
    const unsigned offW0 = (unsigned)(n_base + rsel) * K2 + ksel;
    const unsigned offW1 = offW0 + (unsigned)Dn * K2;
    const unsigned offW2 = offW1 + (unsigned)Dn * K2;
    const unsigned offW3 = offW2 + (unsigned)Dn * K2;

    v8f c00 = {0.f,0.f,0.f,0.f,0.f,0.f,0.f,0.f};
    v8f c01 = c00, c10 = c00, c11 = c00, c20 = c00, c21 = c00, c30 = c00, c31 = c00;

#pragma unroll 2
    for (int k0 = 0; k0 < K2; k0 += 32) {
        __builtin_prefetch(A + offA0 + k0 + 1024, 0, 1);      // ~2KB ahead
        // Load order: first WMMA pair depends only on the first 4 loads.
        v16bf w0 = load_frag(W, offW0 + k0);
        v16bf a0 = load_frag(A, offA0 + k0);
        v16bf a1 = load_frag(A, offA1 + k0);
        v16bf w1 = load_frag(W, offW1 + k0);
        v16bf w2 = load_frag(W, offW2 + k0);
        v16bf w3 = load_frag(W, offW3 + k0);
        // Only VMEM reads (0x20) may be scheduled across this point.
        __builtin_amdgcn_sched_barrier(0x20);
        c00 = WMMA_BF16(a0, w0, c00);
        c01 = WMMA_BF16(a1, w0, c01);
        c10 = WMMA_BF16(a0, w1, c10);
        c11 = WMMA_BF16(a1, w1, c11);
        c20 = WMMA_BF16(a0, w2, c20);
        c21 = WMMA_BF16(a1, w2, c21);
        c30 = WMMA_BF16(a0, w3, c30);
        c31 = WMMA_BF16(a1, w3, c31);
    }

    // ---------------- Epilogue: per-lane elementwise LSTM update ------------
    // C/D layout: VGPR r, lane L -> M = r + 8*(L>>4), N = L&15.
    const int lhalf = lane >> 4;
    const int col   = n_base + rsel;
    const float bf_ = bsum[col];
    const float bi_ = bsum[Dn + col];
    const float bo_ = bsum[2 * Dn + col];
    const float bc_ = bsum[3 * Dn + col];

#define EPILOGUE8(CF, CI, CO, CC, MROW0)                                       \
    do {                                                                       \
        _Pragma("unroll")                                                      \
        for (int r = 0; r < 8; ++r) {                                          \
            const int    m   = (MROW0) + r;                                    \
            const size_t off = (size_t)m * Dn + col;                           \
            const float fg = sigm ((CF)[r] + bf_);                             \
            const float ig = sigm ((CI)[r] + bi_);                             \
            const float og = sigm ((CO)[r] + bo_);                             \
            const float cg = tanhp((CC)[r] + bc_);                             \
            const float cn = fg * c_in[off] + ig * cg;                         \
            const float hn = og * cn;                                          \
            out_o[off] = og;                                                   \
            out_h[off] = hn;                                                   \
            out_c[off] = cn;                                                   \
        }                                                                      \
    } while (0)

    EPILOGUE8(c00, c10, c20, c30, m0 + (lhalf << 3));
    EPILOGUE8(c01, c11, c21, c31, m0 + 16 + (lhalf << 3));
#undef EPILOGUE8
}

// ---------------------------------------------------------------------------
extern "C" void kernel_launch(void* const* d_in, const int* in_sizes, int n_in,
                              void* d_out, int out_size, void* d_ws, size_t ws_size,
                              hipStream_t stream) {
    const float* x   = (const float*)d_in[0];
    const float* h   = (const float*)d_in[1];
    const float* c   = (const float*)d_in[2];
    const float* Wfx = (const float*)d_in[3];
    const float* bfx = (const float*)d_in[4];
    const float* Wfh = (const float*)d_in[5];
    const float* bfh = (const float*)d_in[6];
    const float* Wix = (const float*)d_in[7];
    const float* bix = (const float*)d_in[8];
    const float* Wih = (const float*)d_in[9];
    const float* bih = (const float*)d_in[10];
    const float* Wox = (const float*)d_in[11];
    const float* box_= (const float*)d_in[12];
    const float* Woh = (const float*)d_in[13];
    const float* boh = (const float*)d_in[14];
    const float* Wcx = (const float*)d_in[15];
    const float* bcx = (const float*)d_in[16];
    const float* Wch = (const float*)d_in[17];
    const float* bch = (const float*)d_in[18];

    // Workspace: A bf16 (64MB) | Wp bf16 (64MB) | bsum f32 (32KB)
    unsigned short* Apk  = (unsigned short*)d_ws;
    unsigned short* Wpk  = Apk + (size_t)Bn * K2;
    float*          bsum = (float*)(Wpk + (size_t)(4 * Dn) * K2);

    float* out   = (float*)d_out;
    float* out_o = out;
    float* out_h = out + (size_t)Bn * Dn;
    float* out_c = out + 2 * (size_t)Bn * Dn;

    const size_t nthreads = (size_t)Bn * K2 / 4;          // 4 elems / thread
    pack_a_kernel<<<dim3((unsigned)(nthreads / 256)), dim3(256), 0, stream>>>(x, h, Apk);
    pack_w_kernel<<<dim3((unsigned)(nthreads / 256)), dim3(256), 0, stream>>>(
        Wfx, Wfh, Wix, Wih, Wox, Woh, Wcx, Wch, Wpk);
    pack_bias_kernel<<<dim3((4 * Dn) / 256), dim3(256), 0, stream>>>(
        bfx, bfh, bix, bih, box_, boh, bcx, bch, bsum);

    dim3 grid(Dn / 16, Bn / 256);
    lstm_gemm_fused<<<grid, dim3(256), 0, stream>>>(Apk, Wpk, bsum, c, out_o, out_h, out_c);
}
